// QuantMultiheadAttention_45664092291657
// MI455X (gfx1250) — compile-verified, matrix operand order checked by reading, not run on hardware
//
#include <hip/hip_runtime.h>

// ---------------------------------------------------------------------------
// QuantMultiheadAttention for MI455X (gfx1250, wave32, WMMA bf16)
// T=2048, B=2, E=1024, H=16, d=64
// ---------------------------------------------------------------------------

typedef unsigned short u16;
typedef __attribute__((ext_vector_type(16))) __bf16 v16bf;
typedef __attribute__((ext_vector_type(8)))  float  v8f;
typedef __attribute__((ext_vector_type(4)))  int    v4i;

union FragU { v16bf v; uint4 q[2]; };

#define WMMA_BF16(a, b, c) \
  __builtin_amdgcn_wmma_f32_16x16x32_bf16(false, (a), false, (b), (short)0, (c), false, false)

// ---- CDNA5 async global->LDS DMA (guarded: fall back to register path) ----
#if defined(__has_builtin)
#if __has_builtin(__builtin_amdgcn_global_load_async_to_lds_b128)
#define HAS_ASYNC_LDS 1
#endif
#endif
#ifndef HAS_ASYNC_LDS
#define HAS_ASYNC_LDS 0
#endif

__device__ __forceinline__ void cp16(const u16* g, u16* l) {
#if HAS_ASYNC_LDS
  __builtin_amdgcn_global_load_async_to_lds_b128(
      (__attribute__((address_space(1))) v4i*)(g),
      (__attribute__((address_space(3))) v4i*)(l), 0, 0);
#else
  *(uint4*)l = *(const uint4*)g;
#endif
}

#if HAS_ASYNC_LDS
#define WAIT_ASYNC() asm volatile("s_wait_asynccnt 0" ::: "memory")
#else
#define WAIT_ASYNC()
#endif

__device__ __forceinline__ u16 f2bf(float f) {
  unsigned u = __float_as_uint(f);
  u += 0x7fffu + ((u >> 16) & 1u);   // round-to-nearest-even
  return (u16)(u >> 16);
}

// Load a 16x32 bf16 fragment (A layout; also used for B from N-major/K-contig
// storage). Lane l: row = l&15; K chunks [k0, k0+8) and [k0+16, k0+24),
// k0 = (l>>4)*8.  base points at (row0, col0) of the fragment; stride in elems.
__device__ __forceinline__ v16bf frag_ld(const u16* base, int stride, int lane) {
  int r  = lane & 15;
  int k0 = (lane >> 4) << 3;
  FragU u;
  u.q[0] = *(const uint4*)(base + r * stride + k0);
  u.q[1] = *(const uint4*)(base + r * stride + k0 + 16);
  return u.v;
}

__device__ __forceinline__ float rmax16(float x) {
  x = fmaxf(x, __shfl_xor(x, 1, 32));
  x = fmaxf(x, __shfl_xor(x, 2, 32));
  x = fmaxf(x, __shfl_xor(x, 4, 32));
  x = fmaxf(x, __shfl_xor(x, 8, 32));
  return x;  // reduced within each 16-lane half (row of the C layout)
}
__device__ __forceinline__ float rsum16(float x) {
  x += __shfl_xor(x, 1, 32);
  x += __shfl_xor(x, 2, 32);
  x += __shfl_xor(x, 4, 32);
  x += __shfl_xor(x, 8, 32);
  return x;
}

// ---------------------------------------------------------------------------
// Fake-quantize fp32 weights -> bf16:  round(clip(w*127/s,-127,127)) * s/127
// ---------------------------------------------------------------------------
__global__ void quant_to_bf16(const float* __restrict__ w, u16* __restrict__ o,
                              int n, float qmul, float dmul) {
  int i = blockIdx.x * 256 + threadIdx.x;
  if (i < n) {
    float x = w[i] * qmul;
    x = fminf(fmaxf(x, -127.f), 127.f);
    o[i] = f2bf(rintf(x) * dmul);
  }
}

__global__ void cvt_to_bf16(const float* __restrict__ x, u16* __restrict__ o, int n) {
  int i = blockIdx.x * 256 + threadIdx.x;
  if (i < n) o[i] = f2bf(x[i]);
}

// ---------------------------------------------------------------------------
// Tiled bf16 GEMM: C[M,N] = A[M,K] * Bw[N,K]^T (+bias). 128x128 block tile,
// 8 waves (2x4), each wave 64x32 = 4x2 WMMA frags. K-tile = 32, double-
// buffered LDS staged by async global->LDS DMA.
// mode 0: QKV epilogue -> scatter bf16 q (*d^-0.5) / k / v into [BH][T][64]
// mode 1: fp32 epilogue -> outF[M,N]
// ---------------------------------------------------------------------------
#define GSTR 40   // K-tile 32 + 8 pad; row byte stride 80 (16B aligned)

__global__ __launch_bounds__(256) void gemm_bf16(
    const u16* __restrict__ A, const u16* __restrict__ Bw,
    const float* __restrict__ bias, int K, int mode,
    float* __restrict__ outF,
    u16* __restrict__ qb, u16* __restrict__ kb, u16* __restrict__ vb) {
  __shared__ u16 At[2][128 * GSTR];
  __shared__ u16 Bt[2][128 * GSTR];

  int tid = threadIdx.x, lane = tid & 31, wid = tid >> 5;
  int M0 = blockIdx.y * 128, N0 = blockIdx.x * 128;
  int waveM = wid >> 2, waveN = wid & 3;

  v8f acc[4][2] = {};

  // staging role: each thread DMAs 2x16B of A and 2x16B of B per K-tile
  int r  = tid >> 1;
  int ch = (tid & 1) * 16;
  const u16* gA = A  + (size_t)(M0 + r) * K + ch;
  const u16* gB = Bw + (size_t)(N0 + r) * K + ch;
  int soff = r * GSTR + ch;

  // prologue: stage K-tile 0 into buffer 0
  cp16(gA, &At[0][soff]);       cp16(gA + 8, &At[0][soff + 8]);
  cp16(gB, &Bt[0][soff]);       cp16(gB + 8, &Bt[0][soff + 8]);

  int pb = 0;
  for (int kt = 0; kt < K; kt += 32) {
    WAIT_ASYNC();          // this wave's DMA into buffer pb is complete
    __syncthreads();       // everyone's DMA complete; prev readers of nb done
    int nb = pb ^ 1;
    if (kt + 32 < K) {     // kick off DMA for the next K-tile
      const u16* nA = gA + kt + 32;
      const u16* nB = gB + kt + 32;
      cp16(nA, &At[nb][soff]);  cp16(nA + 8, &At[nb][soff + 8]);
      cp16(nB, &Bt[nb][soff]);  cp16(nB + 8, &Bt[nb][soff + 8]);
      if (kt + 64 < K) {
        __builtin_prefetch(gA + kt + 64, 0, 0);
        __builtin_prefetch(gB + kt + 64, 0, 0);
      }
    }

    v16bf bf0 = frag_ld(&Bt[pb][(waveN * 32 +  0) * GSTR], GSTR, lane);
    v16bf bf1 = frag_ld(&Bt[pb][(waveN * 32 + 16) * GSTR], GSTR, lane);
#pragma unroll
    for (int mf = 0; mf < 4; mf++) {
      v16bf af = frag_ld(&At[pb][(waveM * 64 + mf * 16) * GSTR], GSTR, lane);
      acc[mf][0] = WMMA_BF16(af, bf0, acc[mf][0]);
      acc[mf][1] = WMMA_BF16(af, bf1, acc[mf][1]);
    }
    pb = nb;
  }

  // Epilogue. C layout: VGPR v, lanes 0-15: M=v, N=lane; lanes 16-31: M=v+8.
  int half = lane >> 4, ln = lane & 15;
#pragma unroll
  for (int mf = 0; mf < 4; mf++)
#pragma unroll
    for (int nf = 0; nf < 2; nf++)
#pragma unroll
      for (int v = 0; v < 8; v++) {
        int gm = M0 + waveM * 64 + mf * 16 + v + half * 8;
        int gn = N0 + waveN * 32 + nf * 16 + ln;
        float val = acc[mf][nf][v] + bias[gn];
        if (mode == 0) {
          // gm = t*2 + b ; gn in [0,3072): part | head | col
          int part = gn >> 10, e = gn & 1023;
          int h = e >> 6, c = e & 63;
          int b = gm & 1, t = gm >> 1;
          if (part == 0) val *= 0.125f;  // q scaling = d^-0.5
          u16* dst = (part == 0) ? qb : ((part == 1) ? kb : vb);
          dst[((size_t)((b * 16 + h) * 2048 + t)) * 64 + c] = f2bf(val);
        } else {
          outF[(size_t)gm * 1024 + gn] = val;
        }
      }
}

// ---------------------------------------------------------------------------
// Fused flash-style attention. Block = (q-tile of 128 rows) x (one head n).
// 8 waves, each wave owns 16 q rows. s-tile = 64 keys/values, double-buffered:
// K tile via async DMA, V tile loaded to regs + transposed into LDS.
// S = q*k^T (WMMA, K=64), online softmax, P staged via per-wave LDS,
// ctx += P*V (WMMA), V staged transposed so the B-operand loader is uniform.
// ---------------------------------------------------------------------------
__global__ __launch_bounds__(256) void attn_kernel(
    const u16* __restrict__ qb, const u16* __restrict__ kb,
    const u16* __restrict__ vb, u16* __restrict__ ctxbf) {
  __shared__ u16 kt[2][64 * 72];     // [s][d]   (B operand for S)
  __shared__ u16 vt[2][64 * 72];     // [d][s]   (B operand for P*V, transposed)
  __shared__ u16 pt[8 * 16 * 72];    // per-wave P tile [16][64] (stride 72)

  int tid = threadIdx.x, lane = tid & 31, wid = tid >> 5;
  int half = lane >> 4, ln = lane & 15;
  int mt = blockIdx.x, n = blockIdx.y;
  int b = n >> 4, h = n & 15;

  const u16* qbase = qb + (size_t)n * 2048 * 64;
  const u16* kbase = kb + (size_t)n * 2048 * 64;
  const u16* vbase = vb + (size_t)n * 2048 * 64;

  int t0 = mt * 128 + wid * 16;
  v16bf qf0 = frag_ld(qbase + (size_t)t0 * 64 + 0, 64, lane);
  v16bf qf1 = frag_ld(qbase + (size_t)t0 * 64 + 32, 64, lane);

  v8f acc[4] = {};
  float mrow[8], lrow[8];
#pragma unroll
  for (int v = 0; v < 8; v++) { mrow[v] = -3.0e38f; lrow[v] = 0.f; }

  int r = tid >> 2, seg = (tid & 3) * 16;   // tile loader: 64 rows x 64 cols
  u16* pwave = &pt[wid * 16 * 72];
  int ksoff = r * 72 + seg;

  // prologue: stage s-tile 0 into buffer 0
  cp16(kbase + (size_t)r * 64 + seg,     &kt[0][ksoff]);
  cp16(kbase + (size_t)r * 64 + seg + 8, &kt[0][ksoff + 8]);
  {
    const uint4* pv = (const uint4*)(vbase + (size_t)r * 64 + seg);
    union { uint4 q; u16 s[8]; } v0, v1;
    v0.q = pv[0]; v1.q = pv[1];
#pragma unroll
    for (int j = 0; j < 8; j++) {
      vt[0][(seg + j) * 72 + r]     = v0.s[j];
      vt[0][(seg + 8 + j) * 72 + r] = v1.s[j];
    }
  }

  int pb = 0;
  for (int st = 0; st < 2048; st += 64) {
    WAIT_ASYNC();
    __syncthreads();         // buffer pb fully staged, buffer nb free
    int nb = pb ^ 1;
    bool more = (st + 64) < 2048;
    union { uint4 q; u16 s[8]; } nv0, nv1;
    if (more) {              // DMA next K tile; prefetch next V into regs
      const u16* nk = kbase + (size_t)(st + 64 + r) * 64 + seg;
      cp16(nk,     &kt[nb][ksoff]);
      cp16(nk + 8, &kt[nb][ksoff + 8]);
      const uint4* pv = (const uint4*)(vbase + (size_t)(st + 64 + r) * 64 + seg);
      nv0.q = pv[0]; nv1.q = pv[1];
    }

    // S = q * k^T : 16 x 64 per wave
    v8f sfr[4] = {};
#pragma unroll
    for (int nf = 0; nf < 4; nf++) {
      v16bf bk0 = frag_ld(&kt[pb][(nf * 16) * 72 + 0], 72, lane);
      sfr[nf] = WMMA_BF16(qf0, bk0, sfr[nf]);
      v16bf bk1 = frag_ld(&kt[pb][(nf * 16) * 72 + 32], 72, lane);
      sfr[nf] = WMMA_BF16(qf1, bk1, sfr[nf]);
    }

    // online softmax, per row (row id = v + 8*half, data across the 16-lane half)
#pragma unroll
    for (int v = 0; v < 8; v++) {
      float rm = -3.0e38f;
#pragma unroll
      for (int nf = 0; nf < 4; nf++) rm = fmaxf(rm, sfr[nf][v]);
      rm = rmax16(rm);
      float mnew  = fmaxf(mrow[v], rm);
      float alpha = __expf(mrow[v] - mnew);
      float ls = 0.f;
#pragma unroll
      for (int nf = 0; nf < 4; nf++) {
        float p = __expf(sfr[nf][v] - mnew);
        sfr[nf][v] = p;
        ls += p;
      }
      ls = rsum16(ls);
      lrow[v] = lrow[v] * alpha + ls;
      mrow[v] = mnew;
#pragma unroll
      for (int f = 0; f < 4; f++) acc[f][v] = acc[f][v] * alpha;
    }

    // P (f32 C-layout) -> bf16 per-wave LDS tile, then reuse as A operand
#pragma unroll
    for (int nf = 0; nf < 4; nf++)
#pragma unroll
      for (int v = 0; v < 8; v++)
        pwave[(v + half * 8) * 72 + nf * 16 + ln] = f2bf(sfr[nf][v]);
    asm volatile("s_wait_dscnt 0" ::: "memory");

    // ctx += P * V
#pragma unroll
    for (int ks = 0; ks < 2; ks++) {
      v16bf pa = frag_ld(pwave + ks * 32, 72, lane);
#pragma unroll
      for (int f = 0; f < 4; f++) {
        v16bf bv = frag_ld(&vt[pb][(f * 16) * 72 + ks * 32], 72, lane);
        acc[f] = WMMA_BF16(pa, bv, acc[f]);
      }
    }

    // scatter next V tile (transposed) into the free buffer; readers of nb
    // finished before this iteration's top barrier, so no extra sync needed.
    if (more) {
#pragma unroll
      for (int j = 0; j < 8; j++) {
        vt[nb][(seg + j) * 72 + r]     = nv0.s[j];
        vt[nb][(seg + 8 + j) * 72 + r] = nv1.s[j];
      }
    }
    pb = nb;
  }

  // normalize and write ctx as bf16 [t*2+b][h*64 + c]
#pragma unroll
  for (int f = 0; f < 4; f++)
#pragma unroll
    for (int v = 0; v < 8; v++) {
      int t = t0 + v + half * 8;
      float val = acc[f][v] * (1.0f / lrow[v]);
      ctxbf[((size_t)t * 2 + b) * 1024 + h * 64 + f * 16 + ln] = f2bf(val);
    }
}

// ---------------------------------------------------------------------------
extern "C" void kernel_launch(void* const* d_in, const int* in_sizes, int n_in,
                              void* d_out, int out_size, void* d_ws, size_t ws_size,
                              hipStream_t stream) {
  (void)in_sizes; (void)n_in; (void)out_size; (void)ws_size;
  const float* query = (const float*)d_in[0];   // [2048,2,1024]
  const float* qkvw  = (const float*)d_in[1];   // [3072,1024]
  const float* qkvb  = (const float*)d_in[2];   // [3072]
  const float* ow    = (const float*)d_in[3];   // [1024,1024]
  const float* ob    = (const float*)d_in[4];   // [1024]

  const int E = 1024, M = 4096, BH = 32, T = 2048, D = 64;
  char* ws = (char*)d_ws;
  size_t off = 0;
  auto carve = [&](size_t bytes) {
    void* p = ws + off;
    off = (off + bytes + 255) & ~(size_t)255;
    return p;
  };
  u16* qkvw_bf = (u16*)carve((size_t)3 * E * E * 2);
  u16* ow_bf   = (u16*)carve((size_t)E * E * 2);
  u16* xbf     = (u16*)carve((size_t)M * E * 2);
  u16* qbuf    = (u16*)carve((size_t)BH * T * D * 2);
  u16* kbuf    = (u16*)carve((size_t)BH * T * D * 2);
  u16* vbuf    = (u16*)carve((size_t)BH * T * D * 2);
  u16* ctxbf   = (u16*)carve((size_t)M * E * 2);

  // 1) fake-quantize weights -> bf16
  {
    int n = 3 * E * E;
    float qmul = 127.0f * 55.42562584220407f;  // 127*sqrt(3072)
    quant_to_bf16<<<(n + 255) / 256, 256, 0, stream>>>(qkvw, qkvw_bf, n, qmul, 1.0f / qmul);
  }
  {
    int n = E * E;
    float qmul = 127.0f * 32.0f;               // 127*sqrt(1024)
    quant_to_bf16<<<(n + 255) / 256, 256, 0, stream>>>(ow, ow_bf, n, qmul, 1.0f / qmul);
  }
  // 2) query -> bf16
  {
    int n = M * E;
    cvt_to_bf16<<<(n + 255) / 256, 256, 0, stream>>>(query, xbf, n);
  }
  // 3) QKV projection GEMM (M=4096, N=3072, K=1024), scatter q/k/v
  {
    dim3 grid(3072 / 128, 4096 / 128);
    gemm_bf16<<<grid, 256, 0, stream>>>(xbf, qkvw_bf, qkvb, E, 0,
                                        nullptr, qbuf, kbuf, vbuf);
  }
  // 4) fused attention -> ctx bf16
  {
    dim3 grid(T / 128, BH);
    attn_kernel<<<grid, 256, 0, stream>>>(qbuf, kbuf, vbuf, ctxbf);
  }
  // 5) output projection GEMM (M=4096, N=1024, K=1024) -> fp32 out
  {
    dim3 grid(1024 / 128, 4096 / 128);
    gemm_bf16<<<grid, 256, 0, stream>>>(ctxbf, ow_bf, ob, E, 1,
                                        (float*)d_out, nullptr, nullptr, nullptr);
  }
}
